// MultiHeadAttention_81758997447243
// MI455X (gfx1250) — matrix operand order repaired and checked by reading, not compiled, not run on hardware
//
#include <hip/hip_runtime.h>
#include <hip/hip_bf16.h>

// ---------------------------------------------------------------------------
// MI455X (gfx1250) multi-head attention forward.
// bf16 WMMA for all matmuls, async global->LDS DMA for all pure-copy staging.
// ---------------------------------------------------------------------------

typedef __bf16 bf16_t;
typedef __attribute__((ext_vector_type(16))) __bf16 vbf16;
typedef __attribute__((ext_vector_type(8)))  __bf16 vbf8;
typedef __attribute__((ext_vector_type(4)))  __bf16 vbf4;
typedef __attribute__((ext_vector_type(8)))  float  vf8;
typedef __attribute__((ext_vector_type(4)))  float  vf4;

union FragU { vbf16 v; vbf8 h[2]; };

// A fragment (16x32 bf16, MxK): lane<16 -> row=lane, k {0..7, 16..23};
// lane>=16 -> row=lane-16, k {8..15, 24..31}.  base points at [row0][k0].
__device__ __forceinline__ vbf16 load_frag_a(const bf16_t* base, int ld, int lane) {
  const int lr = lane & 15, hi = lane >> 4;
  const bf16_t* p = base + (size_t)lr * ld + hi * 8;
  FragU r;
  r.h[0] = *(const vbf8*)(p);
  r.h[1] = *(const vbf8*)(p + 16);
  return r.v;
}

// B fragment (32x16 bf16, KxN) read from Bt[n][k] (N-major, K contiguous):
// lane<16 -> col n=lane, k {0..7, 16..23}; lane>=16 -> col n=lane-16, k {8..15, 24..31}.
__device__ __forceinline__ vbf16 load_frag_b(const bf16_t* base, int ld, int lane) {
  const int lr = lane & 15, hi = lane >> 4;
  const bf16_t* p = base + (size_t)lr * ld + hi * 8;
  FragU r;
  r.h[0] = *(const vbf8*)(p);
  r.h[1] = *(const vbf8*)(p + 16);
  return r.v;
}

__device__ __forceinline__ vf8 wmma_bf16(vbf16 a, vbf16 b, vf8 c) {
  return __builtin_amdgcn_wmma_f32_16x16x32_bf16(
      /*neg_a=*/false, a, /*neg_b=*/false, b,
      /*c_mod=*/(short)0, c, /*reuse_a=*/false, /*reuse_b=*/false);
}

// Generic pointer to a __shared__ object is (aperture<<32)|lds_offset on
// amdgcn: low 32 bits are the LDS byte address the async DMA wants.
__device__ __forceinline__ uint32_t lds_u32(const void* p) {
  return (uint32_t)(uintptr_t)p;
}

// Async DMA: 16 bytes global -> LDS per lane, tracked by ASYNCcnt (no VGPR
// data path).  CDNA5 ISA / CDNA5 ISA opcode 98.
__device__ __forceinline__ void async_copy_b128(uint32_t lds_byte_addr,
                                                const void* gptr) {
  asm volatile("global_load_async_to_lds_b128 %0, %1, off"
               :: "v"(lds_byte_addr), "v"(gptr)
               : "memory");
}
__device__ __forceinline__ void async_wait0() {
  asm volatile("s_wait_asynccnt 0x0" ::: "memory");
}

// ---------------------------------------------------------------------------
// Kernel 0: W (KxN fp32, row-major) -> Wt (NxK bf16, row-major)
// ---------------------------------------------------------------------------
__global__ __launch_bounds__(256) void transpose_f32_to_bf16(
    const float* __restrict__ W, bf16_t* __restrict__ Wt, int K, int N) {
  __shared__ float tile[32][33];
  const int n0 = blockIdx.x * 32, k0 = blockIdx.y * 32;
  const int tx = threadIdx.x, ty = threadIdx.y;  // (32, 8)
#pragma unroll
  for (int i = 0; i < 32; i += 8)
    tile[ty + i][tx] = W[(size_t)(k0 + ty + i) * N + n0 + tx];
  __syncthreads();
#pragma unroll
  for (int i = 0; i < 32; i += 8)
    Wt[(size_t)(n0 + ty + i) * K + k0 + tx] = (bf16_t)tile[tx][ty + i];
}

// ---------------------------------------------------------------------------
// GEMM: C(MxN) = A(MxK) * Bt(NxK)^T + bias.  BM=BN=128, BK=64.
// 256 threads = 8 waves as 4(m) x 2(n); each wave owns 32x64 (2x4 WMMA tiles).
// A_F32: fp32 activations converted to bf16 on the LDS staging path;
//        otherwise A is bf16 and staged via async DMA.
// EPI: 0 = bf16 row-major, 1 = fp32 row-major (d_out),
//      2 = bf16 per-head transposed V layout Vt[b][h][d][s] (for flash PV).
// ---------------------------------------------------------------------------
template <bool A_F32, int EPI>
__global__ __launch_bounds__(256) void gemm_wmma(
    const void* __restrict__ Av, const bf16_t* __restrict__ Bt,
    const float* __restrict__ bias, void* __restrict__ Cv,
    int M, int N, int K) {
  __shared__ bf16_t lA[128][72];  // row stride 144B (16B aligned)
  __shared__ bf16_t lB[128][72];
  const int t = threadIdx.x;
  const int lane = t & 31, wid = t >> 5;
  const int wm = wid & 3, wn = wid >> 2;
  const int m0 = blockIdx.y * 128, n0 = blockIdx.x * 128;
  const uint32_t lA0 = lds_u32(&lA[0][0]);
  const uint32_t lB0 = lds_u32(&lB[0][0]);

  vf8 acc[2][4] = {};

  for (int kb = 0; kb < K; kb += 64) {
    {  // B tile: bf16 copy -> async DMA straight into LDS
      const int row = t >> 3, c8 = (t & 7) * 8;
#pragma unroll
      for (int p = 0; p < 4; ++p)
        async_copy_b128(lB0 + (uint32_t)((row + p * 32) * 72 + c8) * 2,
                        Bt + (size_t)(n0 + row + p * 32) * K + kb + c8);
    }
    if (A_F32) {  // fp32 -> bf16 conversion must pass through VGPRs
      const float* A = (const float*)Av;
      const int row = t >> 4, c4 = (t & 15) * 4;
#pragma unroll
      for (int p = 0; p < 8; ++p) {
        vf4 f = *(const vf4*)(A + (size_t)(m0 + row + p * 16) * K + kb + c4);
        vbf4 bv;
#pragma unroll
        for (int e = 0; e < 4; ++e) bv[e] = (bf16_t)f[e];
        *(vbf4*)&lA[row + p * 16][c4] = bv;
      }
    } else {  // bf16 A: async DMA
      const bf16_t* A = (const bf16_t*)Av;
      const int row = t >> 3, c8 = (t & 7) * 8;
#pragma unroll
      for (int p = 0; p < 4; ++p)
        async_copy_b128(lA0 + (uint32_t)((row + p * 32) * 72 + c8) * 2,
                        A + (size_t)(m0 + row + p * 32) * K + kb + c8);
    }
    async_wait0();
    __syncthreads();

#pragma unroll
    for (int ks = 0; ks < 2; ++ks) {
      vbf16 af[2];
#pragma unroll
      for (int i = 0; i < 2; ++i)
        af[i] = load_frag_a(&lA[wm * 32 + i * 16][ks * 32], 72, lane);
#pragma unroll
      for (int j = 0; j < 4; ++j) {
        vbf16 bfv = load_frag_b(&lB[wn * 64 + j * 16][ks * 32], 72, lane);
#pragma unroll
        for (int i = 0; i < 2; ++i) acc[i][j] = wmma_bf16(af[i], bfv, acc[i][j]);
      }
    }
    __syncthreads();
  }

  const int lr = lane & 15, hi = lane >> 4;
#pragma unroll
  for (int i = 0; i < 2; ++i) {
#pragma unroll
    for (int j = 0; j < 4; ++j) {
      const int col = n0 + wn * 64 + j * 16 + lr;
      const float bv = bias[col];
#pragma unroll
      for (int r = 0; r < 8; ++r) {
        const int row = m0 + wm * 32 + i * 16 + hi * 8 + r;
        const float vv = acc[i][j][r] + bv;
        if (EPI == 1) {
          ((float*)Cv)[(size_t)row * N + col] = vv;
        } else if (EPI == 0) {
          ((bf16_t*)Cv)[(size_t)row * N + col] = (bf16_t)vv;
        } else {
          // V projection: write per-head transposed Vt[b][h][d][s].
          // row = b*2048 + t', col = channel; head h = t'>>7,
          // s = (t'&127)*16 + col/64, d = col&63 (raw-reshape semantics).
          const int b = row >> 11, tp = row & 2047;
          const int h = tp >> 7;
          const int s = ((tp & 127) << 4) + (col >> 6);
          const int d = col & 63;
          ((bf16_t*)Cv)[((size_t)(b * 16 + h) << 17) + ((size_t)d << 11) + s] =
              (bf16_t)vv;
        }
      }
    }
  }
}

// ---------------------------------------------------------------------------
// Kernel 2: causal flash attention per (b, h).  Q/K per head are contiguous
// (2048,64) row-major blocks at element offset (b*2048+h*128)*1024 (raw
// reshape).  V comes pre-transposed per head: Vt[b][h][d][s], ld = 2048.
// Grid: x = query tile (128 rows), y = b*16 + h.  8 waves; wave w owns
// query rows [w*16, w*16+16) and all 128 key columns of the tile.
// scale = 1/sqrt(E) = 1/32 (reference quirk).
// ---------------------------------------------------------------------------
__global__ __launch_bounds__(256) void flash_attn_kernel(
    const bf16_t* __restrict__ Qp, const bf16_t* __restrict__ Kp,
    const bf16_t* __restrict__ Vt, bf16_t* __restrict__ Op) {
  __shared__ bf16_t lK[128][72];     // K tile [s][d]  == Bt layout for Q*K^T
  __shared__ bf16_t lV[64][136];     // V^T tile [d][s] == Bt layout for P*V
  __shared__ bf16_t lP[8][16][136];  // per-wave P tile (C-layout -> A-layout)

  const int t = threadIdx.x, lane = t & 31, wid = t >> 5;
  const int lr = lane & 15, hi = lane >> 4;
  const int bh = blockIdx.y, b = bh >> 4, h = bh & 15;
  const size_t base = (size_t)(b * 2048 + h * 128) * 1024;
  const bf16_t* Qh = Qp + base;
  const bf16_t* Kh = Kp + base;
  const bf16_t* Vh = Vt + ((size_t)bh << 17);  // [d][s], ld 2048
  const int m0 = blockIdx.x * 128;
  const int mrow = m0 + wid * 16;
  const uint32_t lK0 = lds_u32(&lK[0][0]);
  const uint32_t lV0 = lds_u32(&lV[0][0]);

  // Q A-fragments straight from global (row-major, ld = 64), two k-steps.
  vbf16 qf[2];
#pragma unroll
  for (int ks = 0; ks < 2; ++ks)
    qf[ks] = load_frag_a(Qh + (size_t)mrow * 64 + ks * 32, 64, lane);

  vf8 o_acc[4] = {};
  float m_run[8], l_run[8];
#pragma unroll
  for (int r = 0; r < 8; ++r) { m_run[r] = -1e30f; l_run[r] = 0.f; }

  const int nst = blockIdx.x + 1;  // causal: only key tiles with s0 <= m0
  for (int st = 0; st < nst; ++st) {
    const int s0 = st * 128;
    {  // K tile: contiguous rows, async DMA into LDS
      const int row = t >> 3, c8 = (t & 7) * 8;
#pragma unroll
      for (int p = 0; p < 4; ++p)
        async_copy_b128(lK0 + (uint32_t)((row + p * 32) * 72 + c8) * 2,
                        Kh + (size_t)(s0 + row + p * 32) * 64 + c8);
    }
    {  // V^T tile: contiguous rows of Vt, async DMA into LDS
      const int row = t >> 4, c8 = (t & 15) * 8;
#pragma unroll
      for (int p = 0; p < 4; ++p)
        async_copy_b128(lV0 + (uint32_t)((row + p * 16) * 136 + c8) * 2,
                        Vh + (size_t)(row + p * 16) * 2048 + s0 + c8);
    }
    async_wait0();
    __syncthreads();

    // S = Q K^T  (16 x 128 per wave)
    vf8 s_acc[8] = {};
#pragma unroll
    for (int j = 0; j < 8; ++j) {
#pragma unroll
      for (int ks = 0; ks < 2; ++ks) {
        vbf16 bfv = load_frag_b(&lK[j * 16][ks * 32], 72, lane);
        s_acc[j] = wmma_bf16(qf[ks], bfv, s_acc[j]);
      }
    }

    // scale + causal mask + online softmax (rows live in 16-lane halves)
#pragma unroll
    for (int r = 0; r < 8; ++r) {
      const int tg = mrow + hi * 8 + r;
      float rmax = -1e30f;
#pragma unroll
      for (int j = 0; j < 8; ++j) {
        const int sg = s0 + j * 16 + lr;
        float vsc = s_acc[j][r] * 0.03125f;  // 1/sqrt(1024)
        if (sg > tg) vsc = -1e30f;
        s_acc[j][r] = vsc;
        rmax = fmaxf(rmax, vsc);
      }
#pragma unroll
      for (int off = 8; off >= 1; off >>= 1)
        rmax = fmaxf(rmax, __shfl_xor(rmax, off, 32));
      const float mnew = fmaxf(m_run[r], rmax);
      const float alpha = __expf(m_run[r] - mnew);
      float rsum = 0.f;
#pragma unroll
      for (int j = 0; j < 8; ++j) {
        const float pv = __expf(s_acc[j][r] - mnew);
        rsum += pv;
        lP[wid][hi * 8 + r][j * 16 + lr] = (bf16_t)pv;
      }
#pragma unroll
      for (int off = 8; off >= 1; off >>= 1)
        rsum += __shfl_xor(rsum, off, 32);
      l_run[r] = l_run[r] * alpha + rsum;
      m_run[r] = mnew;
#pragma unroll
      for (int jn = 0; jn < 4; ++jn) o_acc[jn][r] *= alpha;
    }

    // O += P V   (A-frags of P from per-wave LDS, B-frags from lV)
#pragma unroll
    for (int ks = 0; ks < 4; ++ks) {
      vbf16 pf = load_frag_a(&lP[wid][0][ks * 32], 136, lane);
#pragma unroll
      for (int jn = 0; jn < 4; ++jn) {
        vbf16 bfv = load_frag_b(&lV[jn * 16][ks * 32], 136, lane);
        o_acc[jn] = wmma_bf16(pf, bfv, o_acc[jn]);
      }
    }
    __syncthreads();
  }

  // write O in (b*T + t, h*64 + d) layout (== reference transpose+reshape)
  bf16_t* Orow = Op + (size_t)(b * 2048) * 1024 + h * 64;
#pragma unroll
  for (int jn = 0; jn < 4; ++jn) {
#pragma unroll
    for (int r = 0; r < 8; ++r) {
      const int tg = mrow + hi * 8 + r;
      const float vv = o_acc[jn][r] / l_run[r];
      Orow[(size_t)tg * 1024 + jn * 16 + lr] = (bf16_t)vv;
    }
  }
}

// ---------------------------------------------------------------------------
extern "C" void kernel_launch(void* const* d_in, const int* in_sizes, int n_in,
                              void* d_out, int out_size, void* d_ws, size_t ws_size,
                              hipStream_t stream) {
  const float* q  = (const float*)d_in[0];
  const float* k  = (const float*)d_in[1];
  const float* v  = (const float*)d_in[2];
  const float* Wq = (const float*)d_in[3];
  const float* bq = (const float*)d_in[4];
  const float* Wk = (const float*)d_in[5];
  const float* bk = (const float*)d_in[6];
  const float* Wv = (const float*)d_in[7];
  const float* bv = (const float*)d_in[8];
  const float* Wo = (const float*)d_in[9];
  const float* bo = (const float*)d_in[10];
  float* out = (float*)d_out;

  const int E = 1024, HD = 1024, M = 4 * 2048;  // B*T rows
  const size_t W_ELE = (size_t)E * HD;          // 1M elems per weight
  const size_t M_ELE = (size_t)M * HD;          // 8M elems per activation

  bf16_t* ws  = (bf16_t*)d_ws;                  // ~75.5 MB of ws used
  bf16_t* Wtq = ws;
  bf16_t* Wtk = Wtq + W_ELE;
  bf16_t* Wtv = Wtk + W_ELE;
  bf16_t* Wto = Wtv + W_ELE;
  bf16_t* Qp  = Wto + W_ELE;
  bf16_t* Kp  = Qp + M_ELE;
  bf16_t* Vtp = Kp + M_ELE;   // per-head transposed V
  bf16_t* Oa  = Vtp + M_ELE;

  dim3 tb(32, 8), tg(E / 32, HD / 32);
  transpose_f32_to_bf16<<<tg, tb, 0, stream>>>(Wq, Wtq, E, HD);
  transpose_f32_to_bf16<<<tg, tb, 0, stream>>>(Wk, Wtk, E, HD);
  transpose_f32_to_bf16<<<tg, tb, 0, stream>>>(Wv, Wtv, E, HD);
  transpose_f32_to_bf16<<<tg, tb, 0, stream>>>(Wo, Wto, HD, E);

  dim3 gg(HD / 128, M / 128);  // (8, 64)
  gemm_wmma<true, 0><<<gg, 256, 0, stream>>>(q, Wtq, bq, Qp, M, HD, E);
  gemm_wmma<true, 0><<<gg, 256, 0, stream>>>(k, Wtk, bk, Kp, M, HD, E);
  gemm_wmma<true, 2><<<gg, 256, 0, stream>>>(v, Wtv, bv, Vtp, M, HD, E);

  dim3 fg(2048 / 128, 4 * 16);  // (qtiles=16, B*H=64)
  flash_attn_kernel<<<fg, 256, 0, stream>>>(Qp, Kp, Vtp, Oa);

  gemm_wmma<false, 1><<<gg, 256, 0, stream>>>(Oa, Wto, bo, out, M, E, HD);
}
